// StandardMultiHeadAttention_62947040690414
// MI455X (gfx1250) — matrix operand order repaired and checked by reading, not compiled, not run on hardware
//
#include <hip/hip_runtime.h>
#include <hip/hip_bf16.h>

typedef __attribute__((ext_vector_type(16))) __bf16 v16bf;
typedef __attribute__((ext_vector_type(8)))  __bf16 v8bf;
typedef __attribute__((ext_vector_type(4)))  __bf16 v4bf;
typedef __attribute__((ext_vector_type(8)))  float  v8f;
typedef __attribute__((ext_vector_type(4)))  float  v4f;

#define D_MODEL 1024
#define NHEAD   16
#define HDIM    64
#define SEQ     2048
#define BATCH   2
#define NTOK    (BATCH*SEQ)   // 4096

// One wave per block; let the register allocator use the full wave32 VGPR file
// (single-wave workgroups hide latency via many blocks per WGP, not waves/SIMD).
#define WMMA_KERNEL_ATTRS \
  __launch_bounds__(32, 1) \
  __attribute__((amdgpu_flat_work_group_size(32, 32), amdgpu_waves_per_eu(1)))

// 1/sqrt(HDIM) * log2(e): softmax evaluated in base-2 via v_exp_f32
#define QSCALE 0.1803368801111204f

__global__ void cvt_f32_to_bf16(const float* __restrict__ src, __bf16* __restrict__ dst, int n4) {
  const v4f* __restrict__ s = (const v4f*)src;
  v4bf* __restrict__ d = (v4bf*)dst;
  int i = blockIdx.x * blockDim.x + threadIdx.x;
  int stride = gridDim.x * blockDim.x;
  for (; i < n4; i += stride) {
    v4f v = s[i];
    v4bf o;
    #pragma unroll
    for (int j = 0; j < 4; ++j) o[j] = (__bf16)v[j];
    d[i] = o;
  }
}

__device__ __forceinline__ v8f vzero8() {
  v8f z;
  #pragma unroll
  for (int i = 0; i < 8; ++i) z[i] = 0.0f;
  return z;
}

// A-operand fragment (16x32 bf16, MxK): lane holds one row, 16 contiguous K values:
// lanes 0-15 hold K=0..15, lanes 16-31 hold K=16..31
__device__ __forceinline__ v16bf load_a16(const __bf16* p) {
  v8bf lo = *(const v8bf*)p;
  v8bf hi = *(const v8bf*)(p + 8);
  v16bf a;
  #pragma unroll
  for (int i = 0; i < 8; ++i) { a[i] = lo[i]; a[8 + i] = hi[i]; }
  return a;
}

// B-operand fragment (32x16 bf16, KxN): lane = column, 16 contiguous K values,
// lanes 0-15 hold K=0..15, lanes 16-31 hold K=16..31 (column-contiguous memory)
__device__ __forceinline__ v16bf load_b16(const __bf16* p) {
  v8bf lo = *(const v8bf*)p;
  v8bf hi = *(const v8bf*)(p + 8);
  v16bf b;
  #pragma unroll
  for (int i = 0; i < 8; ++i) { b[i] = lo[i]; b[8 + i] = hi[i]; }
  return b;
}

// C[m][n] = sum_k A[m0+m][k] * W[n0+n][k]  (A @ W^T), 64x64 tile per wave:
// 16 WMMAs per 8 fragment loads per k-step; cacheline-ahead prefetch on both streams.
__device__ __forceinline__ void gemm64x64(const __bf16* __restrict__ A,
                                          const __bf16* __restrict__ W,
                                          int m0, int n0, int Kdim, v8f acc[4][4]) {
  int lane = threadIdx.x & 31;
  int rlo = lane & 15, half = lane >> 4;
  const __bf16* ap[4];
  const __bf16* bp[4];
  #pragma unroll
  for (int t = 0; t < 4; ++t) {
    ap[t] = A + (size_t)(m0 + 16*t + rlo)*Kdim + half*16;
    bp[t] = W + (size_t)(n0 + 16*t + rlo)*Kdim + half*16;
  }
  for (int k0 = 0; k0 < Kdim; k0 += 32) {
    v16bf a[4];
    #pragma unroll
    for (int mt = 0; mt < 4; ++mt) {
      a[mt] = load_a16(ap[mt] + k0);
      __builtin_prefetch(ap[mt] + k0 + 64, 0, 3);   // global_prefetch_b8, 1 line ahead
    }
    #pragma unroll
    for (int nt = 0; nt < 4; ++nt) {
      v16bf b = load_b16(bp[nt] + k0);
      __builtin_prefetch(bp[nt] + k0 + 64, 0, 3);
      #pragma unroll
      for (int mt = 0; mt < 4; ++mt)
        acc[mt][nt] = __builtin_amdgcn_wmma_f32_16x16x32_bf16(
            false, a[mt], false, b, (short)0, acc[mt][nt], false, false);
    }
  }
}

// mode 0: Q (scaled, [bh][s][dh]); mode 1: K ([bh][s][dh]); mode 2: V transposed ([bh][dh][s])
__global__ WMMA_KERNEL_ATTRS
void qkv_proj_kernel(const __bf16* __restrict__ X, const __bf16* __restrict__ W,
                     const float* __restrict__ bias, __bf16* __restrict__ out,
                     int mode) {
  int bid = blockIdx.x;
  int m0 = (bid % (NTOK/64)) * 64;
  int n0 = (bid / (NTOK/64)) * 64;
  v8f acc[4][4];
  #pragma unroll
  for (int mt = 0; mt < 4; ++mt)
    #pragma unroll
    for (int nt = 0; nt < 4; ++nt) acc[mt][nt] = vzero8();

  gemm64x64(X, W, m0, n0, D_MODEL, acc);

  int lane = threadIdx.x & 31;
  int nlo = lane & 15, half = lane >> 4;
  #pragma unroll
  for (int nt = 0; nt < 4; ++nt) {
    int col = n0 + 16*nt + nlo;
    float bv = bias[col];
    int h = col >> 6, dh = col & 63;
    #pragma unroll
    for (int mt = 0; mt < 4; ++mt) {
      #pragma unroll
      for (int r = 0; r < 8; ++r) {
        int m = m0 + 16*mt + r + 8*half;
        int bidx = m >> 11, s = m & (SEQ - 1);
        float v = acc[mt][nt][r] + bv;
        size_t idx;
        if (mode == 2) {
          idx = ((size_t)(bidx*NHEAD + h)*HDIM + dh)*SEQ + s;      // V^T
        } else {
          if (mode == 0) v *= QSCALE;
          idx = ((size_t)(bidx*NHEAD + h)*SEQ + s)*HDIM + dh;      // Q / K
        }
        out[idx] = (__bf16)v;
      }
    }
  }
}

__global__ WMMA_KERNEL_ATTRS
void out_proj_kernel(const __bf16* __restrict__ O, const __bf16* __restrict__ W,
                     const float* __restrict__ bias, float* __restrict__ out) {
  int bid = blockIdx.x;
  int m0 = (bid % (NTOK/64)) * 64;
  int n0 = (bid / (NTOK/64)) * 64;
  v8f acc[4][4];
  #pragma unroll
  for (int mt = 0; mt < 4; ++mt)
    #pragma unroll
    for (int nt = 0; nt < 4; ++nt) acc[mt][nt] = vzero8();

  gemm64x64(O, W, m0, n0, D_MODEL, acc);

  int lane = threadIdx.x & 31;
  int nlo = lane & 15, half = lane >> 4;
  #pragma unroll
  for (int nt = 0; nt < 4; ++nt) {
    int col = n0 + 16*nt + nlo;
    float bv = bias[col];
    #pragma unroll
    for (int mt = 0; mt < 4; ++mt) {
      #pragma unroll
      for (int r = 0; r < 8; ++r) {
        int m = m0 + 16*mt + r + 8*half;
        out[(size_t)m*D_MODEL + col] = acc[mt][nt][r] + bv;  // coalesced b32 across lanes
      }
    }
  }
}

// Flash attention, one wave per (bh, 16-row query block).
// Computes St = K·Q^T (transposed scores) so each lane owns a fixed query column:
// softmax stats are lane-local + one shfl_xor(16); O^T = V^T·P^T via WMMA.
__global__ WMMA_KERNEL_ATTRS
void attn_kernel(const __bf16* __restrict__ Q, const __bf16* __restrict__ Km,
                 const __bf16* __restrict__ Vt, __bf16* __restrict__ O) {
  int lane = threadIdx.x & 31;
  int n = lane & 15, half = lane >> 4;
  int bid = blockIdx.x;
  int qb = bid & (SEQ/16 - 1);
  int bh = bid >> 7;                       // SEQ/16 == 128
  int q0 = qb * 16;
  const __bf16* Qb = Q  + (size_t)bh * SEQ * HDIM;
  const __bf16* Kb = Km + (size_t)bh * SEQ * HDIM;
  const __bf16* Vb = Vt + (size_t)bh * HDIM * SEQ;

  // Q^T as B-operand, K-dim = dh (two 32-chunks)
  v16bf qf[2];
  #pragma unroll
  for (int kk = 0; kk < 2; ++kk)
    qf[kk] = load_b16(Qb + (size_t)(q0 + n)*HDIM + kk*32 + half*16);

  v8f o[4];
  #pragma unroll
  for (int mt = 0; mt < 4; ++mt) o[mt] = vzero8();
  float m_run = -1e30f, l_run = 0.0f;

  int nj = ((q0 + 15) >> 5) + 1;           // causal: kv blocks of 32
  for (int j = 0; j < nj; ++j) {
    int kv0 = j << 5;
    v8f st[2];
    st[0] = vzero8(); st[1] = vzero8();
    #pragma unroll
    for (int t = 0; t < 2; ++t)
      #pragma unroll
      for (int kk = 0; kk < 2; ++kk) {
        v16bf a = load_a16(Kb + (size_t)(kv0 + 16*t + n)*HDIM + kk*32 + half*16);
        st[t] = __builtin_amdgcn_wmma_f32_16x16x32_bf16(
            false, a, false, qf[kk], (short)0, st[t], false, false);
      }

    if (j == nj - 1) {                     // only last block crosses the diagonal
      #pragma unroll
      for (int t = 0; t < 2; ++t)
        #pragma unroll
        for (int r = 0; r < 8; ++r) {
          int kv = kv0 + 16*t + r + 8*half;
          if (kv > q0 + n) st[t][r] = -3.0e38f;
        }
    }

    // column (== per-query) max: 16 lane-local values + cross-half combine
    float mloc = st[0][0];
    #pragma unroll
    for (int r = 1; r < 8; ++r) mloc = fmaxf(mloc, st[0][r]);
    #pragma unroll
    for (int r = 0; r < 8; ++r) mloc = fmaxf(mloc, st[1][r]);
    mloc = fmaxf(mloc, __shfl_xor(mloc, 16, 32));
    float m_new = fmaxf(m_run, mloc);
    float alpha = __builtin_amdgcn_exp2f(m_run - m_new);

    v8f p0, p1;
    float psum = 0.0f;
    #pragma unroll
    for (int r = 0; r < 8; ++r) {
      p0[r] = __builtin_amdgcn_exp2f(st[0][r] - m_new);
      p1[r] = __builtin_amdgcn_exp2f(st[1][r] - m_new);
      psum += p0[r] + p1[r];
    }
    psum += __shfl_xor(psum, 16, 32);
    l_run = l_run * alpha + psum;
    m_run = m_new;
    #pragma unroll
    for (int mt = 0; mt < 4; ++mt) o[mt] *= alpha;

    // P^T into B-operand layout: swap kv 8..15 <-> 16..23 across lane halves
    v16bf pf;
    #pragma unroll
    for (int r = 0; r < 8; ++r) {
      float t0 = p0[r], t1 = p1[r];
      float ot0 = __shfl_xor(t0, 16, 32);
      float ot1 = __shfl_xor(t1, 16, 32);
      pf[r]     = (__bf16)(half ? ot1 : t0);
      pf[8 + r] = (__bf16)(half ? t1  : ot0);
    }

    // O^T += V^T_tile(16 dh rows) x P^T
    #pragma unroll
    for (int mt = 0; mt < 4; ++mt) {
      v16bf a = load_a16(Vb + (size_t)(16*mt + n)*SEQ + kv0 + half*16);
      o[mt] = __builtin_amdgcn_wmma_f32_16x16x32_bf16(
          false, a, false, pf, (short)0, o[mt], false, false);
    }
  }

  float inv = 1.0f / l_run;
  int b = bh >> 4, h = bh & (NHEAD - 1);
  size_t tok = (size_t)b*SEQ + q0 + n;
  #pragma unroll
  for (int mt = 0; mt < 4; ++mt)
    #pragma unroll
    for (int r = 0; r < 8; ++r) {
      int dh = 16*mt + r + 8*half;
      O[tok*D_MODEL + h*HDIM + dh] = (__bf16)(o[mt][r] * inv);
    }
}

extern "C" void kernel_launch(void* const* d_in, const int* in_sizes, int n_in,
                              void* d_out, int out_size, void* d_ws, size_t ws_size,
                              hipStream_t stream) {
  const float* x  = (const float*)d_in[0];
  const float* Wq = (const float*)d_in[1];
  const float* bq = (const float*)d_in[2];
  const float* Wk = (const float*)d_in[3];
  const float* bk = (const float*)d_in[4];
  const float* Wv = (const float*)d_in[5];
  const float* bv = (const float*)d_in[6];
  const float* Wo = (const float*)d_in[7];
  const float* bo = (const float*)d_in[8];
  float* out = (float*)d_out;

  __bf16* ws = (__bf16*)d_ws;
  const size_t NX = (size_t)NTOK * D_MODEL;             // 4M elems
  const size_t NW = (size_t)D_MODEL * D_MODEL;          // 1M elems
  const size_t NH = (size_t)BATCH * NHEAD * SEQ * HDIM; // 4M elems
  __bf16* xbf = ws;  ws += NX;
  __bf16* wqb = ws;  ws += NW;
  __bf16* wkb = ws;  ws += NW;
  __bf16* wvb = ws;  ws += NW;
  __bf16* wob = ws;  ws += NW;
  __bf16* Qb  = ws;  ws += NH;
  __bf16* Kb  = ws;  ws += NH;
  __bf16* Vtb = ws;  ws += NH;
  __bf16* Ob  = ws;  ws += NX;

  cvt_f32_to_bf16<<<1024, 256, 0, stream>>>(x,  xbf, (int)(NX/4));
  cvt_f32_to_bf16<<<256,  256, 0, stream>>>(Wq, wqb, (int)(NW/4));
  cvt_f32_to_bf16<<<256,  256, 0, stream>>>(Wk, wkb, (int)(NW/4));
  cvt_f32_to_bf16<<<256,  256, 0, stream>>>(Wv, wvb, (int)(NW/4));
  cvt_f32_to_bf16<<<256,  256, 0, stream>>>(Wo, wob, (int)(NW/4));

  const int gemmGrid = (NTOK/64) * (D_MODEL/64);        // 1024 single-wave blocks
  qkv_proj_kernel<<<gemmGrid, 32, 0, stream>>>(xbf, wqb, bq, Qb, 0);
  qkv_proj_kernel<<<gemmGrid, 32, 0, stream>>>(xbf, wkb, bk, Kb, 1);
  qkv_proj_kernel<<<gemmGrid, 32, 0, stream>>>(xbf, wvb, bv, Vtb, 2);

  attn_kernel<<<BATCH*NHEAD*(SEQ/16), 32, 0, stream>>>(Qb, Kb, Vtb, Ob);

  out_proj_kernel<<<gemmGrid, 32, 0, stream>>>(Ob, wob, bo, out);
}
